// DGCNN_seg_27058293965186
// MI455X (gfx1250) — compile-verified, matrix operand order checked by reading, not run on hardware
//
#include <hip/hip_runtime.h>

// ---------------- problem constants ----------------
#define NPOINTS 2048
#define NBATCH  8
#define KNBR    40
#define NP      (NBATCH * NPOINTS)   // 16384 points
#define ME      (NP * KNBR)          // 655360 edges

typedef __attribute__((ext_vector_type(16))) _Float16 v16h;
typedef __attribute__((ext_vector_type(8)))  float    v8f;

// ---------------- WMMA helpers ----------------
__device__ __forceinline__ v8f wmma16(v16h a, v16h b, v8f c) {
  return __builtin_amdgcn_wmma_f32_16x16x32_f16(false, a, false, b, (short)0, c,
                                                false, false);
}

// K-position of fragment element e for this lane (16-bit A/B layout, wave32)
__device__ __forceinline__ int kpos(int lane, int e) {
  int v = e >> 1, p = e & 1;
  return ((v >> 2) << 4) + ((lane >> 4) << 3) + ((v & 3) << 1) + p;
}
// base K of element pair v (elements 2v, 2v+1)
__device__ __forceinline__ int kpair(int lane, int v) {
  return ((v >> 2) << 4) + ((lane >> 4) << 3) + ((v & 3) << 1);
}

// Packed B fragment: one aligned 32-byte load per (kc,nt,lane)
__device__ __forceinline__ v16h load_bfrag_packed(const _Float16* __restrict__ Wp,
                                                  int kc, int nt, int NT, int lane) {
  return ((const v16h*)Wp)[(kc * NT + nt) * 32 + lane];
}

// float atomic max via int/uint punning (mixed signs, -big init)
__device__ __forceinline__ void atomicMaxF(float* a, float v) {
  if (v >= 0.f) atomicMax((int*)a, __float_as_int(v));
  else          atomicMin((unsigned int*)a, __float_as_uint(v));
}

// ---------------- utility kernels ----------------
__global__ void fill_kernel(float* __restrict__ p, int n, float v) {
  for (int i = blockIdx.x * blockDim.x + threadIdx.x; i < n;
       i += gridDim.x * blockDim.x)
    p[i] = v;
}

// Pack row-major W into WMMA B-fragment order, f16, zero-padded.
// Packed K rows = HALVES*CP; packed row k -> half=k/CP, kk=k%CP,
// source row = half*CREAL+kk (zero if kk>=CREAL).
__global__ void pack_w_kernel(const float* __restrict__ W,
                              _Float16* __restrict__ out, int CP, int CREAL,
                              int HALVES, int NOUT) {
  int total = HALVES * CP * NOUT;
  int NT = NOUT >> 4;
  for (int i = blockIdx.x * 256 + threadIdx.x; i < total; i += gridDim.x * 256) {
    int e = i & 15;
    int lane = (i >> 4) & 31;
    int tile = i >> 9;                 // kc*NT + nt
    int kc = tile / NT, nt = tile % NT;
    int k = kc * 32 + kpos(lane, e);
    int n = nt * 16 + (lane & 15);
    int half = k / CP, kk = k % CP;
    float v = (kk < CREAL) ? W[(size_t)(half * CREAL + kk) * NOUT + n] : 0.f;
    out[i] = (_Float16)v;
  }
}

// pad positions [NP,3] -> [NP,16] (zero pad)
__global__ void padpos_kernel(const float* __restrict__ pos,
                              float* __restrict__ out) {
  int m = blockIdx.x * 256 + threadIdx.x;
  if (m >= NP) return;
  float* o = out + (size_t)m * 16;
#pragma unroll
  for (int c = 0; c < 16; ++c) o[c] = (c < 3) ? pos[(size_t)m * 3 + c] : 0.f;
}

// ---------------- exact f32 KNN, one block per query (padded stride C) ------
template <int C>
__global__ __launch_bounds__(256) void knn_kernel(const float* __restrict__ x,
                                                  int* __restrict__ out) {
  __shared__ float q[C];
  __shared__ float dist[NPOINTS];
  __shared__ float rv[256];
  __shared__ int   ri[256];
  const int p = blockIdx.x;
  const int b = p >> 11;
  const int i = p & (NPOINTS - 1);
  const int tid = threadIdx.x;
  if (tid < C) q[tid] = x[(size_t)p * C + tid];
  __syncthreads();
  const float* xb = x + (size_t)(b << 11) * C;
  for (int j = tid; j < NPOINTS; j += 256) {
    const float* xj = xb + (size_t)j * C;
    float d = 0.f;
#pragma unroll
    for (int c = 0; c < C; c += 4) {
      float4 a4 = *(const float4*)(xj + c);
      float d0 = q[c] - a4.x, d1 = q[c + 1] - a4.y;
      float d2 = q[c + 2] - a4.z, d3 = q[c + 3] - a4.w;
      d += d0 * d0 + d1 * d1 + d2 * d2 + d3 * d3;
    }
    dist[j] = (j == i) ? 1e30f : d;
  }
  __syncthreads();
  for (int k = 0; k < KNBR; ++k) {
    float bv = 3.0e38f; int bi = NPOINTS;
    for (int j = tid; j < NPOINTS; j += 256) {
      float d = dist[j];
      if (d < bv) { bv = d; bi = j; }
    }
    rv[tid] = bv; ri[tid] = bi;
    __syncthreads();
    for (int s = 128; s > 0; s >>= 1) {
      if (tid < s) {
        float ov = rv[tid + s]; int oi = ri[tid + s];
        if (ov < rv[tid] || (ov == rv[tid] && oi < ri[tid])) { rv[tid] = ov; ri[tid] = oi; }
      }
      __syncthreads();
    }
    if (tid == 0) { out[(size_t)p * KNBR + k] = ri[0]; dist[ri[0]] = 1e30f; }
    __syncthreads();
  }
}

// ---------------- BN finalize ----------------
__global__ void bnfin_kernel(const float* __restrict__ stats,
                             const float* __restrict__ g,
                             const float* __restrict__ be, float count, int C,
                             float* __restrict__ sc, float* __restrict__ sh) {
  int t = blockIdx.x * 256 + threadIdx.x;
  if (t >= C) return;
  float mean = stats[t] / count;
  float var  = stats[C + t] / count - mean * mean;
  float s = g[t] * rsqrtf(var + 1e-5f);
  sc[t] = s;
  sh[t] = be[t] - mean * s;
}

// ---------------- EdgeConv 2-layer MLP, WMMA, packed weights ----------------
// x has padded stride CP; edge K-layout = [x_i(CP) , x_j - x_i(CP)]
// MODE 0: y1 -> stats ; MODE 1: bn1/lrelu -> y2 -> stats ; MODE 2: full + max-K
template <int CP, int COUT1, int COUT2, int MODE>
__global__ __launch_bounds__(256) void ec_kernel(
    const float* __restrict__ x, const int* __restrict__ nbr,
    const _Float16* __restrict__ W1p, const float* __restrict__ B1,
    const float* __restrict__ sc1, const float* __restrict__ sh1,
    const _Float16* __restrict__ W2p, const float* __restrict__ B2,
    const float* __restrict__ sc2, const float* __restrict__ sh2,
    float* __restrict__ stats, float* __restrict__ out) {
  constexpr int NT1 = COUT1 / 16;
  constexpr int NT2 = COUT2 / 16;
  constexpr int KC1 = (2 * CP) / 32;
  constexpr int CMAX = (COUT1 > COUT2) ? COUT1 : COUT2;
  __shared__ _Float16 hbuf[8][16][COUT1];
  __shared__ float ssum[CMAX], ssq[CMAX];

  const int lane = threadIdx.x & 31;
  const int wv   = threadIdx.x >> 5;
  const int rowbase = blockIdx.x * 128 + wv * 16;   // ME % 128 == 0

  if (MODE == 0 || MODE == 1) {
    for (int t = threadIdx.x; t < CMAX; t += 256) { ssum[t] = 0.f; ssq[t] = 0.f; }
    __syncthreads();
  }

  const int m1 = rowbase + (lane & 15);
  const int gi = m1 / KNBR;
  const int bb = m1 / (KNBR * NPOINTS);
  const int gj = bb * NPOINTS + nbr[m1];
  const float* xip = x + (size_t)gi * CP;
  const float* xjp = x + (size_t)gj * CP;

  // ---- layer 1 ----
  v8f acc1[NT1] = {};
#pragma unroll
  for (int kc = 0; kc < KC1; ++kc) {
    v16h a;
#pragma unroll
    for (int v = 0; v < 8; ++v) {
      int k0 = kc * 32 + kpair(lane, v);
      int kk = k0 & (CP - 1);
      float vi0 = xip[kk],     vi1 = xip[kk + 1];
      float vj0 = xjp[kk],     vj1 = xjp[kk + 1];
      bool first = k0 < CP;
      a[2 * v]     = (_Float16)(first ? vi0 : vj0 - vi0);
      a[2 * v + 1] = (_Float16)(first ? vi1 : vj1 - vi1);
    }
#pragma unroll
    for (int nt = 0; nt < NT1; ++nt)
      acc1[nt] = wmma16(a, load_bfrag_packed(W1p, kc, nt, NT1, lane), acc1[nt]);
  }
#pragma unroll
  for (int nt = 0; nt < NT1; ++nt) {
    float bias = B1[nt * 16 + (lane & 15)];
#pragma unroll
    for (int r = 0; r < 8; ++r) acc1[nt][r] += bias;
  }

  if (MODE == 0) {
#pragma unroll
    for (int nt = 0; nt < NT1; ++nt) {
      float s = 0.f, q = 0.f;
#pragma unroll
      for (int r = 0; r < 8; ++r) { float v = acc1[nt][r]; s += v; q += v * v; }
      int n = nt * 16 + (lane & 15);
      atomicAdd(&ssum[n], s); atomicAdd(&ssq[n], q);
    }
    __syncthreads();
    for (int t = threadIdx.x; t < COUT1; t += 256) {
      atomicAdd(&stats[t], ssum[t]);
      atomicAdd(&stats[COUT1 + t], ssq[t]);
    }
    return;
  }

  // bn1 + lrelu, stage to LDS (transpose to A-layout)
#pragma unroll
  for (int nt = 0; nt < NT1; ++nt) {
    int n = nt * 16 + (lane & 15);
    float s = sc1[n], h = sh1[n];
#pragma unroll
    for (int r = 0; r < 8; ++r) {
      float v = acc1[nt][r] * s + h;
      v = (v >= 0.f) ? v : 0.2f * v;
      hbuf[wv][r + ((lane >> 4) << 3)][n] = (_Float16)v;
    }
  }
  __syncthreads();

  // ---- layer 2 ----
  v8f acc2[NT2] = {};
#pragma unroll
  for (int kc = 0; kc < COUT1 / 32; ++kc) {
    v16h a;
#pragma unroll
    for (int e = 0; e < 16; ++e)
      a[e] = hbuf[wv][lane & 15][kc * 32 + kpos(lane, e)];
#pragma unroll
    for (int nt = 0; nt < NT2; ++nt)
      acc2[nt] = wmma16(a, load_bfrag_packed(W2p, kc, nt, NT2, lane), acc2[nt]);
  }
#pragma unroll
  for (int nt = 0; nt < NT2; ++nt) {
    float bias = B2[nt * 16 + (lane & 15)];
#pragma unroll
    for (int r = 0; r < 8; ++r) acc2[nt][r] += bias;
  }

  if (MODE == 1) {
#pragma unroll
    for (int nt = 0; nt < NT2; ++nt) {
      float s = 0.f, q = 0.f;
#pragma unroll
      for (int r = 0; r < 8; ++r) { float v = acc2[nt][r]; s += v; q += v * v; }
      int n = nt * 16 + (lane & 15);
      atomicAdd(&ssum[n], s); atomicAdd(&ssq[n], q);
    }
    __syncthreads();
    for (int t = threadIdx.x; t < COUT2; t += 256) {
      atomicAdd(&stats[t], ssum[t]);
      atomicAdd(&stats[COUT2 + t], ssq[t]);
    }
    return;
  }

  // MODE 2: bn2 + lrelu + max over K neighbors
#pragma unroll
  for (int nt = 0; nt < NT2; ++nt) {
    int n = nt * 16 + (lane & 15);
    float s = sc2[n], h = sh2[n];
#pragma unroll
    for (int r = 0; r < 8; ++r) {
      float v = acc2[nt][r] * s + h;
      v = (v >= 0.f) ? v : 0.2f * v;
      int m = rowbase + r + ((lane >> 4) << 3);
      atomicMaxF(&out[(size_t)(m / KNBR) * COUT2 + n], v);
    }
  }
}

// ---------------- t2: [NP,128] @ W[128,1024], BN, lrelu, max pool ----------
template <int MODE>
__global__ __launch_bounds__(256) void t2_kernel(
    const float* __restrict__ x, const _Float16* __restrict__ Wp,
    const float* __restrict__ Bb, const float* __restrict__ sc,
    const float* __restrict__ sh, float* __restrict__ stats,
    float* __restrict__ pooled) {
  __shared__ float sred[2048];
  const int lane = threadIdx.x & 31, wv = threadIdx.x >> 5;
  const int rowbase = blockIdx.x * 128 + wv * 16;
  if (MODE == 0) {
    for (int t = threadIdx.x; t < 2048; t += 256) sred[t] = 0.f;
  } else {
    for (int t = threadIdx.x; t < 1024; t += 256) sred[t] = -3.0e38f;
  }
  __syncthreads();
  const float* xp = x + (size_t)(rowbase + (lane & 15)) * 128;
  v16h af[4];
#pragma unroll
  for (int kc = 0; kc < 4; ++kc)
#pragma unroll
    for (int v = 0; v < 8; ++v) {
      int k0 = kc * 32 + kpair(lane, v);
      af[kc][2 * v]     = (_Float16)xp[k0];
      af[kc][2 * v + 1] = (_Float16)xp[k0 + 1];
    }

  for (int nt = 0; nt < 64; ++nt) {
    v8f acc = {};
#pragma unroll
    for (int kc = 0; kc < 4; ++kc)
      acc = wmma16(af[kc], load_bfrag_packed(Wp, kc, nt, 64, lane), acc);
    int n = nt * 16 + (lane & 15);
    float bias = Bb[n];
    if (MODE == 0) {
      float s = 0.f, q = 0.f;
#pragma unroll
      for (int r = 0; r < 8; ++r) { float v = acc[r] + bias; s += v; q += v * v; }
      atomicAdd(&sred[n], s); atomicAdd(&sred[1024 + n], q);
    } else {
      float scl = sc[n], shf = sh[n];
      float mx = -3.0e38f;
#pragma unroll
      for (int r = 0; r < 8; ++r) {
        float v = (acc[r] + bias) * scl + shf;
        v = (v >= 0.f) ? v : 0.2f * v;
        mx = fmaxf(mx, v);
      }
      atomicMaxF(&sred[n], mx);
    }
  }
  __syncthreads();
  if (MODE == 0) {
    for (int t = threadIdx.x; t < 1024; t += 256) {
      atomicAdd(&stats[t], sred[t]);
      atomicAdd(&stats[1024 + t], sred[1024 + t]);
    }
  } else {
    int batch = (blockIdx.x * 128) >> 11;
    for (int t = threadIdx.x; t < 1024; t += 256)
      atomicMaxF(&pooled[batch * 1024 + t], sred[t]);
  }
}

// ---------------- transformer head: 1024->512->256->9 ----------------
__global__ __launch_bounds__(256) void t3t4_kernel(
    const float* __restrict__ pooled, const float* __restrict__ w3a,
    const float* __restrict__ b3a, const float* __restrict__ w3b,
    const float* __restrict__ b3b, const float* __restrict__ w4,
    const float* __restrict__ b4, float* __restrict__ T) {
  __shared__ float in0[1024], h1[512], h2[256];
  const int tid = threadIdx.x;
  for (int b = 0; b < NBATCH; ++b) {
    for (int t = tid; t < 1024; t += 256) in0[t] = pooled[b * 1024 + t];
    __syncthreads();
    for (int o = tid; o < 512; o += 256) {
      float s = b3a[o];
      for (int i = 0; i < 1024; ++i) s += in0[i] * w3a[i * 512 + o];
      h1[o] = (s >= 0.f) ? s : 0.2f * s;
    }
    __syncthreads();
    for (int o = tid; o < 256; o += 256) {
      float s = b3b[o];
      for (int i = 0; i < 512; ++i) s += h1[i] * w3b[i * 256 + o];
      h2[o] = (s >= 0.f) ? s : 0.2f * s;
    }
    __syncthreads();
    if (tid < 9) {
      float s = b4[tid];
      for (int i = 0; i < 256; ++i) s += h2[i] * w4[i * 9 + tid];
      T[b * 9 + tid] = s;
    }
    __syncthreads();
  }
}

// ---------------- apply 3x3 transform + concat, write stride 32 ------------
__global__ void transform_kernel(const float* __restrict__ pos,
                                 const float* __restrict__ feat,
                                 const float* __restrict__ T,
                                 float* __restrict__ xout) {
  int m = blockIdx.x * 256 + threadIdx.x;
  if (m >= NP) return;
  const float* t = T + (m >> 11) * 9;
  float p0 = pos[(size_t)m * 3], p1 = pos[(size_t)m * 3 + 1], p2 = pos[(size_t)m * 3 + 2];
  float* o = xout + (size_t)m * 32;
#pragma unroll
  for (int j = 0; j < 3; ++j) o[j] = p0 * t[j] + p1 * t[3 + j] + p2 * t[6 + j];
#pragma unroll
  for (int c = 0; c < 16; ++c) o[3 + c] = feat[(size_t)m * 16 + c];
#pragma unroll
  for (int c = 19; c < 32; ++c) o[c] = 0.f;
}

// ---------------- per-layer head (packed weights, padded x) ----------------
template <int CPX, int KCX>
__global__ __launch_bounds__(256) void head_kernel(
    const float* __restrict__ xi, const float* __restrict__ x,
    const _Float16* __restrict__ W1p, const float* __restrict__ B1,
    const _Float16* __restrict__ W2p, const float* __restrict__ B2,
    const _Float16* __restrict__ WLp, const float* __restrict__ BL,
    float* __restrict__ out) {
  __shared__ _Float16 hbuf[8][16][64];
  const int lane = threadIdx.x & 31, wv = threadIdx.x >> 5;
  const int rowbase = blockIdx.x * 128 + wv * 16;
  const int mrowA = rowbase + (lane & 15);

  // u = relu(xi @ W1 + b1)
  const float* xip = xi + (size_t)mrowA * 64;
  v8f accu[4] = {};
#pragma unroll
  for (int kc = 0; kc < 2; ++kc) {
    v16h a;
#pragma unroll
    for (int v = 0; v < 8; ++v) {
      int k0 = kc * 32 + kpair(lane, v);
      a[2 * v]     = (_Float16)xip[k0];
      a[2 * v + 1] = (_Float16)xip[k0 + 1];
    }
#pragma unroll
    for (int nt = 0; nt < 4; ++nt)
      accu[nt] = wmma16(a, load_bfrag_packed(W1p, kc, nt, 4, lane), accu[nt]);
  }
#pragma unroll
  for (int nt = 0; nt < 4; ++nt) {
    int n = nt * 16 + (lane & 15);
    float bias = B1[n];
#pragma unroll
    for (int r = 0; r < 8; ++r) {
      float v = accu[nt][r] + bias;
      hbuf[wv][r + ((lane >> 4) << 3)][n] = (_Float16)(v > 0.f ? v : 0.f);
    }
  }
  __syncthreads();

  // v = u @ W2
  v8f accv[4] = {};
#pragma unroll
  for (int kc = 0; kc < 2; ++kc) {
    v16h a;
#pragma unroll
    for (int e = 0; e < 16; ++e)
      a[e] = hbuf[wv][lane & 15][kc * 32 + kpos(lane, e)];
#pragma unroll
    for (int nt = 0; nt < 4; ++nt)
      accv[nt] = wmma16(a, load_bfrag_packed(W2p, kc, nt, 4, lane), accv[nt]);
  }

  // r = x @ WL (shortcut; x padded stride CPX, WL packed to CPX rows)
  const float* xp = x + (size_t)mrowA * CPX;
  v8f accr[4] = {};
#pragma unroll
  for (int kc = 0; kc < KCX; ++kc) {
    v16h a;
#pragma unroll
    for (int v = 0; v < 8; ++v) {
      int k0 = kc * 32 + kpair(lane, v);
      a[2 * v]     = (_Float16)xp[k0];
      a[2 * v + 1] = (_Float16)xp[k0 + 1];
    }
#pragma unroll
    for (int nt = 0; nt < 4; ++nt)
      accr[nt] = wmma16(a, load_bfrag_packed(WLp, kc, nt, 4, lane), accr[nt]);
  }

#pragma unroll
  for (int nt = 0; nt < 4; ++nt) {
    int n = nt * 16 + (lane & 15);
    float cb = B2[n] + BL[n];
#pragma unroll
    for (int r = 0; r < 8; ++r) {
      int m = rowbase + r + ((lane >> 4) << 3);
      out[(size_t)m * 64 + n] = accv[nt][r] + accr[nt][r] + cb;
    }
  }
}

// ---------------- host orchestration ----------------
extern "C" void kernel_launch(void* const* d_in, const int* in_sizes, int n_in,
                              void* d_out, int out_size, void* d_ws,
                              size_t ws_size, hipStream_t stream) {
  (void)out_size; (void)ws_size;
  const float* pos;
  const float* feat;
  int pbase;
  bool alpha;
  if (in_sizes[0] == NP * 3) {                 // doc order: pos, feat, bidx, params
    pos = (const float*)d_in[0];
    feat = (const float*)d_in[1];
    pbase = 3;
    alpha = (in_sizes[3] == 38 * 64);
  } else {                                     // fully alphabetized pytree
    feat = (const float*)d_in[1];
    pos = (const float*)d_in[n_in - 1];
    pbase = 2;
    alpha = true;
  }
  const float* P[60];
  for (int i = 0; i < 60; ++i) P[i] = (const float*)d_in[pbase + i];
  int o_conv, o_lin, o_lt, o_t1, o_t2, o_t3, o_t4;
  if (alpha) { o_conv = 0;  o_lin = 24; o_lt = 36; o_t1 = 42; o_t2 = 50; o_t3 = 54; o_t4 = 58; }
  else       { o_t1 = 0;   o_t2 = 8;  o_t3 = 12; o_t4 = 16; o_conv = 18; o_lin = 42; o_lt = 54; }

  const float *t1w1 = P[o_t1+0], *t1b1 = P[o_t1+1], *t1g1 = P[o_t1+2], *t1e1 = P[o_t1+3];
  const float *t1w2 = P[o_t1+4], *t1b2 = P[o_t1+5], *t1g2 = P[o_t1+6], *t1e2 = P[o_t1+7];
  const float *t2w = P[o_t2+0], *t2b = P[o_t2+1], *t2g = P[o_t2+2], *t2e = P[o_t2+3];
  const float *t3w1 = P[o_t3+0], *t3b1 = P[o_t3+1], *t3w2 = P[o_t3+2], *t3b2 = P[o_t3+3];
  const float *t4w = P[o_t4+0], *t4b = P[o_t4+1];
  const float *cw1[3], *cb1[3], *cg1[3], *ce1[3], *cw2[3], *cb2[3], *cg2[3], *ce2[3];
  const float *lw1[3], *lb1[3], *lw2[3], *lb2[3], *ltw[3], *ltb[3];
  for (int l = 0; l < 3; ++l) {
    const float** c = P + o_conv + 8 * l;
    cw1[l] = c[0]; cb1[l] = c[1]; cg1[l] = c[2]; ce1[l] = c[3];
    cw2[l] = c[4]; cb2[l] = c[5]; cg2[l] = c[6]; ce2[l] = c[7];
    const float** li = P + o_lin + 4 * l;
    lw1[l] = li[0]; lb1[l] = li[1]; lw2[l] = li[2]; lb2[l] = li[3];
    const float** lt = P + o_lt + 2 * l;
    ltw[l] = lt[0]; ltb[l] = lt[1];
  }

  // ---- workspace layout (all offsets multiples of 8 floats / 32 bytes) ----
  float* cur = (float*)d_ws;
  auto take = [&](size_t nfl) { float* p = cur; cur += nfl; return p; };
  int*   idxb  = (int*)take(ME);              // ME ints
  float* xT    = take((size_t)NP * 128);
  float* pos16 = take((size_t)NP * 16);
  float* xA32  = take((size_t)NP * 32);
  float* x64a  = take((size_t)NP * 64);
  float* x64b  = take((size_t)NP * 64);
  float* xie   = take((size_t)NP * 64);
  float* pooled = take(8 * 1024);
  float* Tm    = take(80);
  float* st1   = take(2048);
  float* st2   = take(256);
  float* s1s   = take(1024);
  float* s1b   = take(1024);
  float* s2s   = take(1024);
  float* s2b   = take(1024);
  // packed f16 weights (sizes in floats = f16/2)
  _Float16* pw_t1a = (_Float16*)take(32 * 64 / 2);
  _Float16* pw_t1b = (_Float16*)take(64 * 128 / 2);
  _Float16* pw_t2  = (_Float16*)take(128 * 1024 / 2);
  _Float16* pw_c1[3]; _Float16* pw_c2[3];
  _Float16* pw_l1[3]; _Float16* pw_l2[3]; _Float16* pw_lt[3];
  pw_c1[0] = (_Float16*)take(64 * 64 / 2);
  pw_c1[1] = (_Float16*)take(128 * 64 / 2);
  pw_c1[2] = (_Float16*)take(128 * 64 / 2);
  for (int l = 0; l < 3; ++l) pw_c2[l] = (_Float16*)take(64 * 64 / 2);
  for (int l = 0; l < 3; ++l) pw_l1[l] = (_Float16*)take(64 * 64 / 2);
  for (int l = 0; l < 3; ++l) pw_l2[l] = (_Float16*)take(64 * 64 / 2);
  pw_lt[0] = (_Float16*)take(32 * 64 / 2);
  pw_lt[1] = (_Float16*)take(64 * 64 / 2);
  pw_lt[2] = (_Float16*)take(64 * 64 / 2);

  dim3 blk(256);
  const float NEG = -3.0e38f;
  auto pack = [&](const float* W, _Float16* dst, int CP, int CREAL, int HALVES,
                  int NOUT) {
    int total = HALVES * CP * NOUT;
    pack_w_kernel<<<(total + 255) / 256, blk, 0, stream>>>(W, dst, CP, CREAL,
                                                           HALVES, NOUT);
  };

  // ---- pack all weights (f16, fragment layout, zero-padded) ----
  pack(t1w1, pw_t1a, 16, 3, 2, 64);       // edge [xi(16),xj-xi(16)] <- rows 0..5
  pack(t1w2, pw_t1b, 64, 64, 1, 128);
  pack(t2w,  pw_t2, 128, 128, 1, 1024);
  pack(cw1[0], pw_c1[0], 32, 19, 2, 64);  // edge [xi(32),xj-xi(32)] <- rows 0..37
  pack(cw1[1], pw_c1[1], 64, 64, 2, 64);
  pack(cw1[2], pw_c1[2], 64, 64, 2, 64);
  for (int l = 0; l < 3; ++l) pack(cw2[l], pw_c2[l], 64, 64, 1, 64);
  for (int l = 0; l < 3; ++l) pack(lw1[l], pw_l1[l], 64, 64, 1, 64);
  for (int l = 0; l < 3; ++l) pack(lw2[l], pw_l2[l], 64, 64, 1, 64);
  pack(ltw[0], pw_lt[0], 32, 19, 1, 64);
  pack(ltw[1], pw_lt[1], 64, 64, 1, 64);
  pack(ltw[2], pw_lt[2], 64, 64, 1, 64);

  padpos_kernel<<<NP / 256, blk, 0, stream>>>(pos, pos16);

  // ---- spatial transformer branch ----
  knn_kernel<16><<<NP, blk, 0, stream>>>(pos16, idxb);
  fill_kernel<<<1, blk, 0, stream>>>(st1, 128, 0.f);
  ec_kernel<16,64,128,0><<<ME/128, blk, 0, stream>>>(pos16, idxb, pw_t1a, t1b1,
      nullptr, nullptr, pw_t1b, t1b2, nullptr, nullptr, st1, nullptr);
  bnfin_kernel<<<1, blk, 0, stream>>>(st1, t1g1, t1e1, (float)ME, 64, s1s, s1b);
  fill_kernel<<<1, blk, 0, stream>>>(st2, 256, 0.f);
  ec_kernel<16,64,128,1><<<ME/128, blk, 0, stream>>>(pos16, idxb, pw_t1a, t1b1,
      s1s, s1b, pw_t1b, t1b2, nullptr, nullptr, st2, nullptr);
  bnfin_kernel<<<1, blk, 0, stream>>>(st2, t1g2, t1e2, (float)ME, 128, s2s, s2b);
  fill_kernel<<<2048, blk, 0, stream>>>(xT, NP * 128, NEG);
  ec_kernel<16,64,128,2><<<ME/128, blk, 0, stream>>>(pos16, idxb, pw_t1a, t1b1,
      s1s, s1b, pw_t1b, t1b2, s2s, s2b, nullptr, xT);

  fill_kernel<<<8, blk, 0, stream>>>(st1, 2048, 0.f);
  t2_kernel<0><<<NP/128, blk, 0, stream>>>(xT, pw_t2, t2b, nullptr, nullptr, st1, nullptr);
  bnfin_kernel<<<4, blk, 0, stream>>>(st1, t2g, t2e, (float)NP, 1024, s1s, s1b);
  fill_kernel<<<32, blk, 0, stream>>>(pooled, 8 * 1024, NEG);
  t2_kernel<1><<<NP/128, blk, 0, stream>>>(xT, pw_t2, t2b, s1s, s1b, nullptr, pooled);
  t3t4_kernel<<<1, blk, 0, stream>>>(pooled, t3w1, t3b1, t3w2, t3b2, t4w, t4b, Tm);
  transform_kernel<<<NP/256, blk, 0, stream>>>(pos, feat, Tm, xA32);

  // ---- 3 dynamic-graph EdgeConv layers ----
  const float* xcur = xA32;
  for (int i = 0; i < 3; ++i) {
    if (i == 0) knn_kernel<32><<<NP, blk, 0, stream>>>(xcur, idxb);
    else        knn_kernel<64><<<NP, blk, 0, stream>>>(xcur, idxb);
    fill_kernel<<<1, blk, 0, stream>>>(st1, 128, 0.f);
    fill_kernel<<<1, blk, 0, stream>>>(st2, 256, 0.f);
    fill_kernel<<<1024, blk, 0, stream>>>(xie, NP * 64, NEG);
    if (i == 0) {
      ec_kernel<32,64,64,0><<<ME/128, blk, 0, stream>>>(xcur, idxb, pw_c1[0], cb1[0],
          nullptr, nullptr, pw_c2[0], cb2[0], nullptr, nullptr, st1, nullptr);
      bnfin_kernel<<<1, blk, 0, stream>>>(st1, cg1[0], ce1[0], (float)ME, 64, s1s, s1b);
      ec_kernel<32,64,64,1><<<ME/128, blk, 0, stream>>>(xcur, idxb, pw_c1[0], cb1[0],
          s1s, s1b, pw_c2[0], cb2[0], nullptr, nullptr, st2, nullptr);
      bnfin_kernel<<<1, blk, 0, stream>>>(st2, cg2[0], ce2[0], (float)ME, 64, s2s, s2b);
      ec_kernel<32,64,64,2><<<ME/128, blk, 0, stream>>>(xcur, idxb, pw_c1[0], cb1[0],
          s1s, s1b, pw_c2[0], cb2[0], s2s, s2b, nullptr, xie);
    } else {
      ec_kernel<64,64,64,0><<<ME/128, blk, 0, stream>>>(xcur, idxb, pw_c1[i], cb1[i],
          nullptr, nullptr, pw_c2[i], cb2[i], nullptr, nullptr, st1, nullptr);
      bnfin_kernel<<<1, blk, 0, stream>>>(st1, cg1[i], ce1[i], (float)ME, 64, s1s, s1b);
      ec_kernel<64,64,64,1><<<ME/128, blk, 0, stream>>>(xcur, idxb, pw_c1[i], cb1[i],
          s1s, s1b, pw_c2[i], cb2[i], nullptr, nullptr, st2, nullptr);
      bnfin_kernel<<<1, blk, 0, stream>>>(st2, cg2[i], ce2[i], (float)ME, 64, s2s, s2b);
      ec_kernel<64,64,64,2><<<ME/128, blk, 0, stream>>>(xcur, idxb, pw_c1[i], cb1[i],
          s1s, s1b, pw_c2[i], cb2[i], s2s, s2b, nullptr, xie);
    }
    float* xnext = (i == 2) ? (float*)d_out : ((i == 0) ? x64a : x64b);
    if (i == 0)
      head_kernel<32,1><<<NP/128, blk, 0, stream>>>(xie, xcur, pw_l1[0], lb1[0],
          pw_l2[0], lb2[0], pw_lt[0], ltb[0], xnext);
    else
      head_kernel<64,2><<<NP/128, blk, 0, stream>>>(xie, xcur, pw_l1[i], lb1[i],
          pw_l2[i], lb2[i], pw_lt[i], ltb[i], xnext);
    xcur = xnext;
  }
}
